// Attention_85590108275078
// MI455X (gfx1250) — matrix operand order repaired and checked by reading, not compiled, not run on hardware
//
#include <hip/hip_runtime.h>

// ---------------------------------------------------------------------------
// Types for CDNA5 WMMA (wave32): A/B are v16h (8 VGPRs), C/D are v8f (8 VGPRs)
// ---------------------------------------------------------------------------
typedef __attribute__((ext_vector_type(16))) _Float16 v16h;
typedef __attribute__((ext_vector_type(8)))  _Float16 v8h;
typedef __attribute__((ext_vector_type(8)))  float    v8f;

union F16x16 { v16h v; v8h h[2]; };
union H4     { _Float16 h[4]; uint2 u; };
union H2     { _Float16 h[2]; unsigned u; };

__device__ __forceinline__ v8f vzero8() {
  v8f z = {0.f, 0.f, 0.f, 0.f, 0.f, 0.f, 0.f, 0.f};
  return z;
}

__device__ __forceinline__ v8f wmma_f16(v16h a, v16h b, v8f c) {
  // D = A(16x32 f16) * B(32x16 f16) + C(16x16 f32)
  return __builtin_amdgcn_wmma_f32_16x16x32_f16(
      /*neg_a=*/false, a, /*neg_b=*/false, b,
      /*c_mod=*/(short)0, c, /*reuse_a=*/false, /*reuse_b=*/false);
}

// Problem constants
#define BB 8
#define SS 1024
#define DD 2048
#define HH 16
#define DHD 128
#define LDP 40    // A/V LDS pitch in halves: 80B rows -> 20-dword stride, conflict-free b128
#define KDP 136   // K LDS pitch in halves: 272B rows -> 68-dword stride, conflict-free b128

// ---------------------------------------------------------------------------
// Projection GEMM: C_f16[8192,2048] = f16( A_f32[8192,2048] @ W_f32[2048,2048] )
// Block tile 128x128, 8 waves (4 in M x 2 in N), wave tile 32x64 = 8 WMMA accs.
// Software-pipelined: next K-slab's global loads are issued before the WMMA
// section so their latency hides behind the matrix work.
// vtrans != 0 stores output transposed as Vt[B][H][DH][S] for the PV stage.
// ---------------------------------------------------------------------------
__global__ void __launch_bounds__(256) proj_gemm_f16(
    const float* __restrict__ A, const float* __restrict__ W,
    _Float16* __restrict__ C, int vtrans)
{
  __shared__ alignas(16) _Float16 lds_a[128 * LDP];  // A tile: [m=128][k=32]
  __shared__ alignas(16) _Float16 lds_b[128 * LDP];  // W tile transposed: [n=128][k=32]

  const int tid   = threadIdx.x;
  const int mbase = blockIdx.x * 128;
  const int nbase = blockIdx.y * 128;
  const int wid   = tid >> 5;
  const int lane  = tid & 31;
  const int wm    = wid & 3;      // rows wm*32
  const int wn    = wid >> 2;     // cols wn*64
  const int lrow  = lane & 15;
  const int lhi   = lane >> 4;

  // ---- staging registers: 4 float4 for A, 2 pairs of float4 for W ----
  float4 ra[4], rb0[2], rb1[2];

  // A mapping: slot f = tid + 256*i -> row = f>>3 (0..127), c4 = f&7
  // W mapping: slot f = tid + 256*i -> krp = f>>5 (0..15) => k-rows 2*krp, 2*krp+1 ; c4 = f&31
  auto issue_loads = [&](int k0) {
#pragma unroll
    for (int i = 0; i < 4; ++i) {
      const int f = tid + 256 * i, row = f >> 3, c4 = f & 7;
      ra[i] = reinterpret_cast<const float4*>(A + (size_t)(mbase + row) * DD + k0)[c4];
    }
#pragma unroll
    for (int i = 0; i < 2; ++i) {
      const int f = tid + 256 * i, krp = f >> 5, c4 = f & 31;
      rb0[i] = reinterpret_cast<const float4*>(W + (size_t)(k0 + 2 * krp)     * DD + nbase)[c4];
      rb1[i] = reinterpret_cast<const float4*>(W + (size_t)(k0 + 2 * krp + 1) * DD + nbase)[c4];
    }
  };

  auto store_lds = [&]() {
#pragma unroll
    for (int i = 0; i < 4; ++i) {
      const int f = tid + 256 * i, row = f >> 3, c4 = f & 7;
      H4 t;
      t.h[0] = (_Float16)ra[i].x; t.h[1] = (_Float16)ra[i].y;
      t.h[2] = (_Float16)ra[i].z; t.h[3] = (_Float16)ra[i].w;
      *reinterpret_cast<uint2*>(&lds_a[row * LDP + c4 * 4]) = t.u;
    }
#pragma unroll
    for (int i = 0; i < 2; ++i) {
      const int f = tid + 256 * i, krp = f >> 5, c4 = f & 31;
      const int kr = 2 * krp, n = c4 * 4;
      const float* p0 = &rb0[i].x;
      const float* p1 = &rb1[i].x;
#pragma unroll
      for (int j = 0; j < 4; ++j) {  // packed b32 store: two k's for same n
        H2 t; t.h[0] = (_Float16)p0[j]; t.h[1] = (_Float16)p1[j];
        *reinterpret_cast<unsigned*>(&lds_b[(n + j) * LDP + kr]) = t.u;
      }
    }
  };

  v8f acc[2][4];
#pragma unroll
  for (int mt = 0; mt < 2; ++mt)
#pragma unroll
    for (int nt = 0; nt < 4; ++nt) acc[mt][nt] = vzero8();

  issue_loads(0);  // prologue

  for (int k0 = 0; k0 < DD; k0 += 32) {
    store_lds();        // consume staging regs (waits on their loads)
    __syncthreads();
    if (k0 + 32 < DD) {
      issue_loads(k0 + 32);   // in flight during the WMMA section below
      if (k0 + 64 < DD) {     // L2 prefetch one more slab ahead
        __builtin_prefetch(A + (size_t)(mbase + (tid >> 1)) * DD + (k0 + 64) + (tid & 1) * 16, 0, 0);
        __builtin_prefetch(W + (size_t)(k0 + 64 + (tid >> 3)) * DD + nbase + (tid & 7) * 16, 0, 0);
      }
    }

    // ---- wave-level 32x64 x 32 WMMA ----
#pragma unroll
    for (int mt = 0; mt < 2; ++mt) {
      F16x16 af;
      const int ar = wm * 32 + mt * 16 + lrow;
      af.h[0] = *reinterpret_cast<const v8h*>(&lds_a[ar * LDP + lhi * 8]);
      af.h[1] = *reinterpret_cast<const v8h*>(&lds_a[ar * LDP + lhi * 8 + 16]);
#pragma unroll
      for (int nt = 0; nt < 4; ++nt) {
        F16x16 bf;
        const int bc = wn * 64 + nt * 16 + lrow;
        bf.h[0] = *reinterpret_cast<const v8h*>(&lds_b[bc * LDP + lhi * 8]);
        bf.h[1] = *reinterpret_cast<const v8h*>(&lds_b[bc * LDP + lhi * 8 + 16]);
        acc[mt][nt] = wmma_f16(af.v, bf.v, acc[mt][nt]);
      }
    }
    __syncthreads();
  }

  // ---- epilogue: f32 acc -> f16, normal or V-transposed layout ----
#pragma unroll
  for (int mt = 0; mt < 2; ++mt)
#pragma unroll
    for (int nt = 0; nt < 4; ++nt)
#pragma unroll
      for (int j = 0; j < 8; ++j) {
        const int m = mbase + wm * 32 + mt * 16 + j + lhi * 8;
        const int n = nbase + wn * 64 + nt * 16 + lrow;
        const _Float16 val = (_Float16)acc[mt][nt][j];
        if (!vtrans) {
          C[(size_t)m * DD + n] = val;                              // [B*S, D]
        } else {
          const int bb = m >> 10, s  = m & 1023;
          const int hh = n >> 7,  dh = n & 127;
          C[((size_t)((bb * HH + hh) * DHD + dh) << 10) + s] = val; // [B,H,DH,S]
        }
      }
}

// ---------------------------------------------------------------------------
// Flash attention. 8 waves/block all share (b,h): the block cooperatively
// stages each 32-key K-chunk (32x128 f16) and V-chunk (128x32 f16, from the
// transposed Vt) into LDS once, cutting global/L2 traffic 8x. Each wave owns
// one 16-row query tile with online softmax. Software-pipelined staging.
// ---------------------------------------------------------------------------
__global__ void __launch_bounds__(256) attn_f16(
    const _Float16* __restrict__ Q, const _Float16* __restrict__ K,
    const _Float16* __restrict__ Vt, float* __restrict__ out)
{
  __shared__ alignas(16) _Float16 lds_k[32 * KDP];    // K chunk: [kv=32][dh=128]
  __shared__ alignas(16) _Float16 lds_v[128 * LDP];   // V chunk: [dh=128][kv=32]
  __shared__ alignas(16) _Float16 pbuf[8][16 * 32];   // per-wave P re-layout slab

  const int tid  = threadIdx.x;
  const int wid  = tid >> 5;
  const int lane = tid & 31;
  const int lrow = lane & 15;
  const int lhi  = lane >> 4;

  // Block-uniform (b,h); per-wave query tile (8 consecutive qt per block).
  const int wgbase = blockIdx.x * 8;
  const int qt = (wgbase & 63) + wid;
  const int bh = wgbase >> 6;
  const int h  = bh & 15;
  const int b  = bh >> 4;
  const int q0 = qt * 16;

  const float kexp = 0.0883883476483184f * 1.4426950408889634f; // (1/sqrt(128))*log2(e)

  const _Float16* kbase = K + (size_t)b * SS * DD + h * DHD;           // + kv*DD + dh
  const _Float16* vbase = Vt + ((size_t)(b * HH + h) * DHD) * SS;      // + dh*SS + kv

  // ---- staging registers (2 x 16B for K, 2 x 16B for V per thread) ----
  v8h rk[2], rv[2];
  auto issue_stage = [&](int j0) {
#pragma unroll
    for (int i = 0; i < 2; ++i) {
      const int s = tid + 256 * i;                 // K: 512 slots of 8 halves
      const int krow = s >> 4, koff = (s & 15) * 8;
      rk[i] = *reinterpret_cast<const v8h*>(kbase + (size_t)(j0 + krow) * DD + koff);
      const int vrow = s >> 2, voff = (s & 3) * 8; // V: 512 slots of 8 halves
      rv[i] = *reinterpret_cast<const v8h*>(vbase + (size_t)vrow * SS + j0 + voff);
    }
  };
  auto store_stage = [&]() {
#pragma unroll
    for (int i = 0; i < 2; ++i) {
      const int s = tid + 256 * i;
      const int krow = s >> 4, koff = (s & 15) * 8;
      *reinterpret_cast<v8h*>(&lds_k[krow * KDP + koff]) = rk[i];
      const int vrow = s >> 2, voff = (s & 3) * 8;
      *reinterpret_cast<v8h*>(&lds_v[vrow * LDP + voff]) = rv[i];
    }
  };

  // ---- load Q A-fragments once (4 chunks of K=32 over DH=128) ----
  F16x16 qf[4];
  {
    const _Float16* qrow = Q + (size_t)(b * SS + q0 + lrow) * DD + h * DHD;
#pragma unroll
    for (int c = 0; c < 4; ++c) {
      qf[c].h[0] = *reinterpret_cast<const v8h*>(qrow + c * 32 + lhi * 8);
      qf[c].h[1] = *reinterpret_cast<const v8h*>(qrow + c * 32 + lhi * 8 + 16);
    }
  }

  v8f o[8];
#pragma unroll
  for (int nt = 0; nt < 8; ++nt) o[nt] = vzero8();
  float rm[8], rl[8];
#pragma unroll
  for (int j = 0; j < 8; ++j) { rm[j] = -1e30f; rl[j] = 0.f; }

  issue_stage(0);  // prologue

  for (int j0 = 0; j0 < SS; j0 += 32) {
    store_stage();
    __syncthreads();
    if (j0 + 32 < SS) {
      issue_stage(j0 + 32);   // in flight during the compute below
      if (j0 + 64 < SS) {
        __builtin_prefetch(kbase + (size_t)(j0 + 64 + (tid >> 3)) * DD + (tid & 7) * 16, 0, 0);
        __builtin_prefetch(vbase + (size_t)(tid >> 1) * SS + j0 + 64, 0, 0);
      }
    }

    // ---- scores: two 16x16 tiles over this 32-key chunk (from lds_k) ----
    v8f s0 = vzero8(), s1 = vzero8();
#pragma unroll
    for (int c = 0; c < 4; ++c) {
      F16x16 kb0, kb1;
      kb0.h[0] = *reinterpret_cast<const v8h*>(&lds_k[lrow * KDP + c * 32 + lhi * 8]);
      kb0.h[1] = *reinterpret_cast<const v8h*>(&lds_k[lrow * KDP + c * 32 + lhi * 8 + 16]);
      kb1.h[0] = *reinterpret_cast<const v8h*>(&lds_k[(16 + lrow) * KDP + c * 32 + lhi * 8]);
      kb1.h[1] = *reinterpret_cast<const v8h*>(&lds_k[(16 + lrow) * KDP + c * 32 + lhi * 8 + 16]);
      s0 = wmma_f16(qf[c].v, kb0.v, s0);
      s1 = wmma_f16(qf[c].v, kb1.v, s1);
    }

    // ---- online softmax in C-layout (row j(+8) striped over 16-lane groups) ----
#pragma unroll
    for (int j = 0; j < 8; ++j) {
      float mx = fmaxf(s0[j], s1[j]);
#pragma unroll
      for (int d = 8; d >= 1; d >>= 1) mx = fmaxf(mx, __shfl_xor(mx, d));
      const float mn = fmaxf(rm[j], mx);
      const float al = exp2f((rm[j] - mn) * kexp);
      rm[j] = mn;
      const float p0 = exp2f((s0[j] - mn) * kexp);
      const float p1 = exp2f((s1[j] - mn) * kexp);
      float rs = p0 + p1;
#pragma unroll
      for (int d = 8; d >= 1; d >>= 1) rs += __shfl_xor(rs, d);
      rl[j] = rl[j] * al + rs;
#pragma unroll
      for (int nt = 0; nt < 8; ++nt) o[nt][j] *= al;  // rescale accumulators
      const int m = j + lhi * 8;
      pbuf[wid][m * 32 + lrow]      = (_Float16)p0;
      pbuf[wid][m * 32 + 16 + lrow] = (_Float16)p1;
    }
    asm volatile("s_wait_dscnt 0" ::: "memory");  // LDS write -> A-frag read (same wave)

    // ---- re-read P as an A-fragment (16x32 f16) ----
    F16x16 pf;
    pf.h[0] = *reinterpret_cast<const v8h*>(&pbuf[wid][lrow * 32 + lhi * 8]);
    pf.h[1] = *reinterpret_cast<const v8h*>(&pbuf[wid][lrow * 32 + lhi * 8 + 16]);

    // ---- O += P @ V over 8 dh-tiles (from lds_v) ----
#pragma unroll
    for (int nt = 0; nt < 8; ++nt) {
      F16x16 vb;
      vb.h[0] = *reinterpret_cast<const v8h*>(&lds_v[(nt * 16 + lrow) * LDP + lhi * 8]);
      vb.h[1] = *reinterpret_cast<const v8h*>(&lds_v[(nt * 16 + lrow) * LDP + lhi * 8 + 16]);
      o[nt] = wmma_f16(pf.v, vb.v, o[nt]);
    }
    __syncthreads();   // protect lds_k/lds_v before next store_stage
  }

  // ---- normalize and store f32 output [B,S,D] ----
  float inv[8];
#pragma unroll
  for (int j = 0; j < 8; ++j) inv[j] = 1.f / rl[j];
#pragma unroll
  for (int nt = 0; nt < 8; ++nt)
#pragma unroll
    for (int j = 0; j < 8; ++j) {
      const size_t m = (size_t)(b * SS + q0 + j + lhi * 8);
      out[m * DD + h * DHD + nt * 16 + lrow] = o[nt][j] * inv[j];
    }
}

// ---------------------------------------------------------------------------
// Host-side launcher. Inputs: q,k,v [8,1024,2048] f32; Wq,Wk,Wv [2048,2048] f32.
// Workspace: Qf16 (32MB) | Kf16 (32MB) | Vt_f16 (32MB).
// ---------------------------------------------------------------------------
extern "C" void kernel_launch(void* const* d_in, const int* in_sizes, int n_in,
                              void* d_out, int out_size, void* d_ws, size_t ws_size,
                              hipStream_t stream) {
  const float* q  = (const float*)d_in[0];
  const float* k  = (const float*)d_in[1];
  const float* v  = (const float*)d_in[2];
  const float* Wq = (const float*)d_in[3];
  const float* Wk = (const float*)d_in[4];
  const float* Wv = (const float*)d_in[5];
  float* out = (float*)d_out;

  const size_t fsz = (size_t)BB * SS * DD * sizeof(_Float16);  // 32 MiB
  _Float16* Qf = (_Float16*)d_ws;
  _Float16* Kf = (_Float16*)((char*)d_ws + fsz);
  _Float16* Vt = (_Float16*)((char*)d_ws + 2 * fsz);

  dim3 gg(BB * SS / 128, DD / 128);  // 64 x 16
  proj_gemm_f16<<<gg, 256, 0, stream>>>(q, Wq, Qf, 0);
  proj_gemm_f16<<<gg, 256, 0, stream>>>(k, Wk, Kf, 0);
  proj_gemm_f16<<<gg, 256, 0, stream>>>(v, Wv, Vt, 1);

  // 8192 (b,h,qtile) waves / 8 waves per block
  attn_f16<<<BB * HH * (SS / 16) / 8, 256, 0, stream>>>(Qf, Kf, Vt, out);
}
